// NeuralMusic_57853209477182
// MI455X (gfx1250) — matrix operand order repaired and checked by reading, not compiled, not run on hardware
//
#include <hip/hip_runtime.h>
#include <hip/hip_bf16.h>
#include <math.h>

typedef __attribute__((ext_vector_type(16))) _Float16 v16h;
typedef __attribute__((ext_vector_type(8)))  _Float16 v8h;
typedef __attribute__((ext_vector_type(8)))  float    v8f;

static constexpr int NB  = 64;    // batch
static constexpr int NF  = 257;   // freq bins
static constexpr int NAZ = 360;   // angles
static constexpr int CFP = 264;   // padded freq (convs2 K inner)
static constexpr int S2_KD  = 3 * CFP;   // 792
static constexpr int S2_KDP = 800;       // padded to 32
static constexpr int S2_MP  = 272;       // padded out channels

// ---------------------------------------------------------------------------
// Packing kernels: NCHW f32 -> NHWC f16, OIHW f32 -> [CoutP][KDP] f16 (tap-major,
// zero padded so WMMA A-fragments need no guards at all).
// ---------------------------------------------------------------------------
__global__ void pack_x_nhwc(const float* __restrict__ X, _Float16* __restrict__ xh)
{
    const int HW = 128 * 128;
    int i = blockIdx.x * blockDim.x + threadIdx.x;
    if (i >= NB * 8 * HW) return;
    int p = i % HW; int t = i / HW;
    int c = t % 8;  int b = t / 8;
    xh[((size_t)b * HW + p) * 8 + c] = (_Float16)X[i];
}

__global__ void pack_w_conv(const float* __restrict__ w, _Float16* __restrict__ wp,
                            int Cout, int Cin, int CoutP, int KDP)
{
    int i = blockIdx.x * blockDim.x + threadIdx.x;
    if (i >= CoutP * KDP) return;
    int kp = i % KDP, mp = i / KDP;
    float v = 0.f;
    int KD = Cin * 9;
    if (mp < Cout && kp < KD) {
        int tap = kp / Cin, cin = kp - tap * Cin;
        int kh = tap / 3, kw = tap - kh * 3;
        v = w[((mp * Cin + cin) * 3 + kh) * 3 + kw];
    }
    wp[i] = (_Float16)v;
}

__global__ void pack_w_s2(const float* __restrict__ w, _Float16* __restrict__ wp)
{
    // w: (257,257,3) -> wp: (272,800), K = dk*264 + cf, zero padded
    int i = blockIdx.x * blockDim.x + threadIdx.x;
    if (i >= S2_MP * S2_KDP) return;
    int kp = i % S2_KDP, mp = i / S2_KDP;
    float v = 0.f;
    if (mp < NF && kp < S2_KD) {
        int dk = kp / CFP, cf = kp - dk * CFP;
        if (cf < NF) v = w[(mp * NF + cf) * 3 + dk];
    }
    wp[i] = (_Float16)v;
}

__global__ void fill_zero_h(_Float16* __restrict__ p, int n)
{
    int i = blockIdx.x * blockDim.x + threadIdx.x;
    if (i < n) p[i] = (_Float16)0.f;
}

// ---------------------------------------------------------------------------
// Implicit-GEMM 3x3 SAME conv + BN + ReLU. NHWC f16 in/out, f32 accum.
// One wave computes a 16(out-ch) x 64(pixel) strip: one A-fragment pair feeds
// 4 WMMAs (4 accumulators) -> 4x weight-stream amortization.
// ---------------------------------------------------------------------------
template <int CIN, int COUT, int KDP>
__global__ __launch_bounds__(32)
void conv3x3_nhwc_wmma(const _Float16* __restrict__ in, const _Float16* __restrict__ wp,
                       const float* __restrict__ gam, const float* __restrict__ bet,
                       const float* __restrict__ mu,  const float* __restrict__ var,
                       _Float16* __restrict__ out, int H, int W)
{
    constexpr int KD = CIN * 9;
    const int lane = threadIdx.x;
    const int half = lane >> 4, l16 = lane & 15;
    const int kgrp = half << 3, kgrb = half << 4;
    const int mt = blockIdx.x, nt = blockIdx.y, b = blockIdx.z;
    const int m = mt * 16 + l16;
    const _Float16* wr = wp + (size_t)m * KDP;

    int yb4[4], xb4[4];
#pragma unroll
    for (int v = 0; v < 4; ++v) {
        int p = (nt * 4 + v) * 16 + l16;
        yb4[v] = p / W;
        xb4[v] = p - yb4[v] * W;
    }

    auto loadB = [&](int kb, int yv, int xv) -> v8h {
        v8h r = {};
        if (kb < KD) {
            int tap = kb / CIN, cin0 = kb - (kb / CIN) * CIN;
            int kh = tap / 3, kw = tap - kh * 3;
            int yy = yv + kh - 1, xx = xv + kw - 1;
            if (yy >= 0 && yy < H && xx >= 0 && xx < W)
                r = *(const v8h*)(in + (((size_t)b * H + yy) * W + xx) * CIN + cin0);
        }
        return r;
    };

    v8f acc[4] = {{}, {}, {}, {}};
#pragma unroll 4
    for (int kc = 0; kc < KDP; kc += 32) {
        __builtin_prefetch(wr + kc + 128, 0, 1);
        v8h a0 = *(const v8h*)(wr + kc + kgrp);
        v8h a1 = *(const v8h*)(wr + kc + 16 + kgrp);
        v16h af = __builtin_shufflevector(a0, a1, 0,1,2,3,4,5,6,7,8,9,10,11,12,13,14,15);
        int kb0 = kc + kgrp, kb1 = kc + 16 + kgrp;
#pragma unroll
        for (int v = 0; v < 4; ++v) {
            v8h b0 = loadB(kb0, yb4[v], xb4[v]);
            v8h b1 = loadB(kb1, yb4[v], xb4[v]);
            v16h bf = __builtin_shufflevector(b0, b1, 0,1,2,3,4,5,6,7,8,9,10,11,12,13,14,15);
            acc[v] = __builtin_amdgcn_wmma_f32_16x16x32_f16(false, af, false, bf,
                                                            (short)0, acc[v], false, false);
        }
    }

    // BN scale/shift per output row (hoisted out of the 4 column strips)
    float sc[8], sh[8];
#pragma unroll
    for (int j = 0; j < 8; ++j) {
        int mr = mt * 16 + (half << 3) + j;
        if (COUT % 8 == 0 || mr < COUT) {
            float s = rsqrtf(var[mr] + 1e-5f) * gam[mr];
            sc[j] = s;
            sh[j] = bet[mr] - mu[mr] * s;
        } else { sc[j] = 0.f; sh[j] = 0.f; }
    }

#pragma unroll
    for (int v = 0; v < 4; ++v) {
        size_t obase = (((size_t)b * H + yb4[v]) * W + xb4[v]) * COUT;
        if constexpr (COUT % 8 == 0) {
            v8h ov;
#pragma unroll
            for (int j = 0; j < 8; ++j) {
                float val = acc[v][j] * sc[j] + sh[j];
                ov[j] = (_Float16)(val > 0.f ? val : 0.f);
            }
            *(v8h*)(out + obase + mt * 16 + (half << 3)) = ov;
        } else {
#pragma unroll
            for (int j = 0; j < 8; ++j) {
                int mr = mt * 16 + (half << 3) + j;
                if (mr >= COUT) continue;
                float val = acc[v][j] * sc[j] + sh[j];
                out[obase + mr] = (_Float16)(val > 0.f ? val : 0.f);
            }
        }
    }
}

// ---------------------------------------------------------------------------
// 2x2 max pool, NHWC f16
// ---------------------------------------------------------------------------
__global__ void maxpool2_nhwc(const _Float16* __restrict__ in, _Float16* __restrict__ out,
                              int C, int H, int W)
{
    int Ho = H >> 1, Wo = W >> 1;
    size_t total = (size_t)NB * Ho * Wo * C;
    size_t i = (size_t)blockIdx.x * blockDim.x + threadIdx.x;
    if (i >= total) return;
    int c = (int)(i % C); size_t t = i / C;
    int xo = (int)(t % Wo); t /= Wo;
    int yo = (int)(t % Ho); int b = (int)(t / Ho);
    const _Float16* ip = in + (((size_t)b * H + 2 * yo) * W + 2 * xo) * C + c;
    size_t rs = (size_t)W * C;
    float v0 = (float)ip[0], v1 = (float)ip[C];
    float v2 = (float)ip[rs], v3 = (float)ip[rs + C];
    out[i] = (_Float16)fmaxf(fmaxf(v0, v1), fmaxf(v2, v3));
}

// ---------------------------------------------------------------------------
// FC: gathers (8,8,257) NHWC -> (B,257,32) f32
// ---------------------------------------------------------------------------
__global__ void fc_kernel(const _Float16* __restrict__ zin, const float* __restrict__ fw,
                          const float* __restrict__ fb, float* __restrict__ zout)
{
    int i = blockIdx.x * blockDim.x + threadIdx.x;
    if (i >= NB * NF * 32) return;
    int o = i & 31, bf = i >> 5;
    int f = bf % NF, b = bf / NF;
    const _Float16* zp = zin + (size_t)b * 64 * NF + f;
    const float* wpt = fw + o * 64;
    float s = fb[o];
    for (int k = 0; k < 64; ++k) s += (float)zp[(size_t)k * NF] * wpt[k];
    zout[i] = s;
}

// ---------------------------------------------------------------------------
// Per (b,f): hermitianize R, complex Jacobi eigh, noise projector UnUn^H.
// ---------------------------------------------------------------------------
__global__ void music_eigh(const float* __restrict__ z, float* __restrict__ unun)
{
    int idx = blockIdx.x * blockDim.x + threadIdx.x;
    if (idx >= NB * NF) return;
    const float* zp = z + (size_t)idx * 32;
    float Ar[4][4], Ai[4][4], Vr[4][4], Vi[4][4];
    for (int i = 0; i < 4; ++i)
        for (int j = 0; j < 4; ++j) {
            float rij = zp[i * 8 + j],     iij = zp[i * 8 + 4 + j];
            float rji = zp[j * 8 + i],     iji = zp[j * 8 + 4 + i];
            Ar[i][j] = 0.5f * (rij + rji);
            Ai[i][j] = 0.5f * (iij - iji);
            Vr[i][j] = (i == j) ? 1.f : 0.f;
            Vi[i][j] = 0.f;
        }
    for (int i = 0; i < 4; ++i) { Ar[i][i] += 1e-5f; Ai[i][i] = 0.f; }

    const int PP[6] = {0, 0, 0, 1, 1, 2}, QQ[6] = {1, 2, 3, 2, 3, 3};
    for (int sw = 0; sw < 8; ++sw)
        for (int pi = 0; pi < 6; ++pi) {
            int pp = PP[pi], q = QQ[pi];
            float ar = Ar[pp][q], ai = Ai[pp][q];
            float mag = sqrtf(ar * ar + ai * ai);
            if (mag < 1e-12f) continue;
            float er = ar / mag, ei = ai / mag;
            float tau = (Ar[q][q] - Ar[pp][pp]) / (2.f * mag);
            float tt = (tau >= 0.f ? 1.f : -1.f) / (fabsf(tau) + sqrtf(1.f + tau * tau));
            float c = 1.f / sqrtf(1.f + tt * tt);
            float s = tt * c;
            for (int r = 0; r < 4; ++r) {
                if (r == pp || r == q) continue;
                float xr = Ar[r][pp], xi = Ai[r][pp];
                float yr = Ar[r][q],  yi = Ai[r][q];
                float cyr = er * yr + ei * yi, cyi = er * yi - ei * yr;
                float exr = er * xr - ei * xi, exi = er * xi + ei * xr;
                float nr0 = c * xr - s * cyr, ni0 = c * xi - s * cyi;
                float nr1 = s * exr + c * yr, ni1 = s * exi + c * yi;
                Ar[r][pp] = nr0;  Ai[r][pp] = ni0;
                Ar[pp][r] = nr0;  Ai[pp][r] = -ni0;
                Ar[r][q]  = nr1;  Ai[r][q]  = ni1;
                Ar[q][r]  = nr1;  Ai[q][r]  = -ni1;
            }
            Ar[pp][pp] -= tt * mag;
            Ar[q][q]   += tt * mag;
            Ar[pp][q] = Ai[pp][q] = Ar[q][pp] = Ai[q][pp] = 0.f;
            for (int r = 0; r < 4; ++r) {
                float xr = Vr[r][pp], xi = Vi[r][pp];
                float yr = Vr[r][q],  yi = Vi[r][q];
                float cyr = er * yr + ei * yi, cyi = er * yi - ei * yr;
                float exr = er * xr - ei * xi, exi = er * xi + ei * xr;
                Vr[r][pp] = c * xr - s * cyr;  Vi[r][pp] = c * xi - s * cyi;
                Vr[r][q]  = s * exr + c * yr;  Vi[r][q]  = s * exi + c * yi;
            }
        }
    float dv[4]; for (int i = 0; i < 4; ++i) dv[i] = Ar[i][i];
    int ord[4] = {0, 1, 2, 3};
    for (int i = 0; i < 3; ++i)
        for (int j = i + 1; j < 4; ++j)
            if (dv[ord[j]] < dv[ord[i]]) { int tsw = ord[i]; ord[i] = ord[j]; ord[j] = tsw; }
    float* up = unun + (size_t)idx * 32;
    for (int mi = 0; mi < 4; ++mi)
        for (int ni = 0; ni < 4; ++ni) {
            float rr = 0.f, ii = 0.f;
            for (int kk = 0; kk < 2; ++kk) {
                int col = ord[kk];
                float arr = Vr[mi][col], aii = Vi[mi][col];
                float brr = Vr[ni][col], bii = Vi[ni][col];
                rr += arr * brr + aii * bii;
                ii += aii * brr - arr * bii;
            }
            up[(mi * 4 + ni) * 2 + 0] = rr;
            up[(mi * 4 + ni) * 2 + 1] = ii;
        }
}

// ---------------------------------------------------------------------------
// MUSIC spectrum -> specT (B, 360, 264) f16 (cf-contiguous, padded with zeros)
// ---------------------------------------------------------------------------
__global__ void music_spectrum(const float* __restrict__ unun,
                               const float* __restrict__ svr,
                               const float* __restrict__ svi,
                               _Float16* __restrict__ specT)
{
    __shared__ float U[32];
    int bf = blockIdx.x;
    int cf = bf % NF, b = bf / NF;
    for (int i = threadIdx.x; i < 32; i += blockDim.x) U[i] = unun[(size_t)bf * 32 + i];
    __syncthreads();
    for (int a = threadIdx.x; a < NAZ; a += blockDim.x) {
        const float* sr = svr + ((size_t)bf * NAZ + a) * 4;
        const float* si = svi + ((size_t)bf * NAZ + a) * 4;
        float cr[4], ci[4];
#pragma unroll
        for (int mm = 0; mm < 4; ++mm) { cr[mm] = sr[mm]; ci[mm] = -si[mm]; }
        float denom = 0.f;
#pragma unroll
        for (int mm = 0; mm < 4; ++mm) {
            float tr = 0.f, ti = 0.f;
#pragma unroll
            for (int nn = 0; nn < 4; ++nn) {
                float ur = U[(mm * 4 + nn) * 2 + 0], ui = U[(mm * 4 + nn) * 2 + 1];
                tr += ur * cr[nn] - ui * ci[nn];
                ti += ur * ci[nn] + ui * cr[nn];
            }
            denom += cr[mm] * tr - ci[mm] * ti;
        }
        float sp = 1.f / fmaxf(denom, 1e-6f);
        specT[((size_t)b * NAZ + a) * CFP + cf] = (_Float16)sp;
    }
}

// ---------------------------------------------------------------------------
// convs2 GEMM: M=257(out), N=360(taps), K=792(padded). bias + leaky ReLU.
// 16 x 64 strip per wave (4 accumulators, shared A).
// ---------------------------------------------------------------------------
__global__ __launch_bounds__(32)
void convs2_wmma(const _Float16* __restrict__ specT, const _Float16* __restrict__ wp,
                 const float* __restrict__ bias, float* __restrict__ out)
{
    const int lane = threadIdx.x;
    const int half = lane >> 4, l16 = lane & 15;
    const int kgrp = half << 3, kgrb = half << 4;
    const int mt = blockIdx.x, nt = blockIdx.y, b = blockIdx.z;
    const int m = mt * 16 + l16;
    const _Float16* wr = wp + (size_t)m * S2_KDP;

    int t4[4];
#pragma unroll
    for (int v = 0; v < 4; ++v) t4[v] = (nt * 4 + v) * 16 + l16;

    auto loadB = [&](int kb, int tv) -> v8h {
        v8h r = {};
        if (kb < S2_KD) {
            int dk = kb / CFP, cf0 = kb - dk * CFP;
            int ttp = tv + dk - 1;
            if (tv < NAZ && ttp >= 0 && ttp < NAZ)
                r = *(const v8h*)(specT + ((size_t)b * NAZ + ttp) * CFP + cf0);
        }
        return r;
    };

    v8f acc[4] = {{}, {}, {}, {}};
#pragma unroll 5
    for (int kc = 0; kc < S2_KDP; kc += 32) {
        __builtin_prefetch(wr + kc + 128, 0, 1);
        v8h a0 = *(const v8h*)(wr + kc + kgrp);
        v8h a1 = *(const v8h*)(wr + kc + 16 + kgrp);
        v16h af = __builtin_shufflevector(a0, a1, 0,1,2,3,4,5,6,7,8,9,10,11,12,13,14,15);
        int kb0 = kc + kgrp, kb1 = kc + 16 + kgrp;
#pragma unroll
        for (int v = 0; v < 4; ++v) {
            v8h b0 = loadB(kb0, t4[v]);
            v8h b1 = loadB(kb1, t4[v]);
            v16h bf = __builtin_shufflevector(b0, b1, 0,1,2,3,4,5,6,7,8,9,10,11,12,13,14,15);
            acc[v] = __builtin_amdgcn_wmma_f32_16x16x32_f16(false, af, false, bf,
                                                            (short)0, acc[v], false, false);
        }
    }
#pragma unroll
    for (int v = 0; v < 4; ++v) {
#pragma unroll
        for (int j = 0; j < 8; ++j) {
            int mr = mt * 16 + (half << 3) + j;
            int col = (nt * 4 + v) * 16 + l16;
            if (mr >= NF || col >= NAZ) continue;
            float val = acc[v][j] + bias[mr];
            val = val > 0.f ? val : 0.1f * val;
            out[((size_t)b * NF + mr) * NAZ + col] = val;
        }
    }
}

// ---------------------------------------------------------------------------
// Channel attention + final conv1d + sigmoid + top-5 soft-argmax
// ---------------------------------------------------------------------------
__global__ void att_mean(const float* __restrict__ spec2, float* __restrict__ y)
{
    int bf = blockIdx.x * blockDim.x + threadIdx.x;
    if (bf >= NB * NF) return;
    const float* sp = spec2 + (size_t)bf * NAZ;
    float s = 0.f;
    for (int t = 0; t < NAZ; ++t) s += sp[t];
    y[bf] = s * (1.0f / NAZ);
}

__global__ void att_mlp(const float* __restrict__ y, const float* __restrict__ w1,
                        const float* __restrict__ w2, float* __restrict__ att)
{
    __shared__ float ys[NF];
    __shared__ float hs[16];
    int b = blockIdx.x;
    for (int f = threadIdx.x; f < NF; f += blockDim.x) ys[f] = y[b * NF + f];
    __syncthreads();
    if (threadIdx.x < 16) {
        float s = 0.f;
        for (int f = 0; f < NF; ++f) s += w1[threadIdx.x * NF + f] * ys[f];
        hs[threadIdx.x] = s > 0.f ? s : 0.f;
    }
    __syncthreads();
    for (int f = threadIdx.x; f < NF; f += blockDim.x) {
        float s = 0.f;
#pragma unroll
        for (int o = 0; o < 16; ++o) s += w2[f * 16 + o] * hs[o];
        att[b * NF + f] = 1.f / (1.f + expf(-s));
    }
}

__global__ void convs1_final(const float* __restrict__ spec2, const float* __restrict__ att,
                             const float* __restrict__ w, const float* __restrict__ bias,
                             float* __restrict__ outspec)
{
    int i = blockIdx.x * blockDim.x + threadIdx.x;
    if (i >= NB * NAZ) return;
    int t = i % NAZ, b = i / NAZ;
    float s = bias[0];
    for (int f = 0; f < NF; ++f) {
        float a = att[b * NF + f];
        const float* sp = spec2 + ((size_t)b * NF + f) * NAZ;
#pragma unroll
        for (int dk = 0; dk < 3; ++dk) {
            int tt = t + dk - 1;
            if (tt >= 0 && tt < NAZ) s += w[f * 3 + dk] * sp[tt] * a;
        }
    }
    outspec[i] = 1.f / (1.f + expf(-s));
}

__global__ void top5_doa(const float* __restrict__ spec, float* __restrict__ doa)
{
    int b = blockIdx.x * blockDim.x + threadIdx.x;
    if (b >= NB) return;
    const float* sp = spec + b * NAZ;
    float vals[5]; int ids[5];
    for (int k = 0; k < 5; ++k) {
        float best = -1e30f; int bi = 0;
        for (int a = 0; a < NAZ; ++a) {
            bool used = false;
            for (int j = 0; j < k; ++j) used = used || (ids[j] == a);
            float v = sp[a];
            if (!used && v > best) { best = v; bi = a; }
        }
        vals[k] = best; ids[k] = bi;
    }
    float mx = vals[0], wsum = 0.f, accum = 0.f;
    for (int k = 0; k < 5; ++k) {
        float e = expf(20.f * (vals[k] - mx));
        wsum += e; accum += e * (float)ids[k];
    }
    doa[b] = accum / wsum;
}

// ---------------------------------------------------------------------------
extern "C" void kernel_launch(void* const* d_in, const int* in_sizes, int n_in,
                              void* d_out, int out_size, void* d_ws, size_t ws_size,
                              hipStream_t stream)
{
    (void)in_sizes; (void)n_in; (void)out_size; (void)ws_size;
    const float* X        = (const float*)d_in[0];
    const float* svr      = (const float*)d_in[1];
    const float* svi      = (const float*)d_in[2];
    const float* conv1_w  = (const float*)d_in[4];
    const float* conv2_w  = (const float*)d_in[5];
    const float* conv3_w  = (const float*)d_in[6];
    const float* conv4_w  = (const float*)d_in[7];
    const float* fc_w     = (const float*)d_in[8];
    const float* fc_b     = (const float*)d_in[9];
    const float* convs2_w = (const float*)d_in[10];
    const float* convs2_b = (const float*)d_in[11];
    const float* convs1_w = (const float*)d_in[12];
    const float* convs1_b = (const float*)d_in[13];
    const float* ca_w1    = (const float*)d_in[14];
    const float* ca_w2    = (const float*)d_in[15];
    const float* bn[4][4];
    for (int i = 0; i < 4; ++i)
        for (int j = 0; j < 4; ++j)
            bn[i][j] = (const float*)d_in[16 + i * 4 + j];

    const size_t MB = 1ull << 20;
    char* ws = (char*)d_ws;
    _Float16* xh   = (_Float16*)(ws);             // 16 MB  (B,128,128,8) f16
    _Float16* wp1  = (_Float16*)(ws + 16 * MB);   // 64 x 96
    _Float16* wp2  = (_Float16*)(ws + 17 * MB);   // 64 x 576
    _Float16* wp3  = (_Float16*)(ws + 18 * MB);   // 64 x 576
    _Float16* wp4  = (_Float16*)(ws + 19 * MB);   // 272 x 576
    _Float16* wps2 = (_Float16*)(ws + 20 * MB);   // 272 x 800
    float*    yb   = (float*)   (ws + 21 * MB);
    float*    att  = (float*)   (ws + 22 * MB);
    char* regA = ws + 24 * MB;     // 136 MB
    char* regB = ws + 160 * MB;    // 40 MB
    char* regC = ws + 200 * MB;    // 40 MB
    float* doa = (float*)d_out;
    float* outspec = doa + NB;

    dim3 w32(32);

    // pack input + weights
    pack_x_nhwc<<<(NB * 8 * 128 * 128 + 255) / 256, 256, 0, stream>>>(X, xh);
    pack_w_conv<<<(64 * 96 + 255) / 256, 256, 0, stream>>>(conv1_w, wp1, 64, 8, 64, 96);
    pack_w_conv<<<(64 * 576 + 255) / 256, 256, 0, stream>>>(conv2_w, wp2, 64, 64, 64, 576);
    pack_w_conv<<<(64 * 576 + 255) / 256, 256, 0, stream>>>(conv3_w, wp3, 64, 64, 64, 576);
    pack_w_conv<<<(272 * 576 + 255) / 256, 256, 0, stream>>>(conv4_w, wp4, 257, 64, 272, 576);
    pack_w_s2<<<(S2_MP * S2_KDP + 255) / 256, 256, 0, stream>>>(convs2_w, wps2);

    // encoder (grids: y = H*W/64 pixel strips)
    conv3x3_nhwc_wmma<8, 64, 96><<<dim3(4, 256, NB), w32, 0, stream>>>(
        xh, wp1, bn[0][0], bn[0][1], bn[0][2], bn[0][3], (_Float16*)regA, 128, 128);
    { size_t n = (size_t)NB * 64 * 64 * 64;
      maxpool2_nhwc<<<(unsigned)((n + 255) / 256), 256, 0, stream>>>(
          (const _Float16*)regA, (_Float16*)regB, 64, 128, 128); }

    conv3x3_nhwc_wmma<64, 64, 576><<<dim3(4, 64, NB), w32, 0, stream>>>(
        (const _Float16*)regB, wp2, bn[1][0], bn[1][1], bn[1][2], bn[1][3],
        (_Float16*)regC, 64, 64);
    { size_t n = (size_t)NB * 32 * 32 * 64;
      maxpool2_nhwc<<<(unsigned)((n + 255) / 256), 256, 0, stream>>>(
          (const _Float16*)regC, (_Float16*)regA, 64, 64, 64); }

    conv3x3_nhwc_wmma<64, 64, 576><<<dim3(4, 16, NB), w32, 0, stream>>>(
        (const _Float16*)regA, wp3, bn[2][0], bn[2][1], bn[2][2], bn[2][3],
        (_Float16*)regB, 32, 32);
    { size_t n = (size_t)NB * 16 * 16 * 64;
      maxpool2_nhwc<<<(unsigned)((n + 255) / 256), 256, 0, stream>>>(
          (const _Float16*)regB, (_Float16*)regC, 64, 32, 32); }

    conv3x3_nhwc_wmma<64, 257, 576><<<dim3(17, 4, NB), w32, 0, stream>>>(
        (const _Float16*)regC, wp4, bn[3][0], bn[3][1], bn[3][2], bn[3][3],
        (_Float16*)regA, 16, 16);
    { size_t n = (size_t)NB * 8 * 8 * NF;
      maxpool2_nhwc<<<(unsigned)((n + 255) / 256), 256, 0, stream>>>(
          (const _Float16*)regA, (_Float16*)regB, NF, 16, 16); }

    // fc -> z (B,257,32) f32
    fc_kernel<<<(NB * NF * 32 + 255) / 256, 256, 0, stream>>>(
        (const _Float16*)regB, fc_w, fc_b, (float*)regC);

    // eigh + noise projector
    music_eigh<<<(NB * NF + 127) / 128, 128, 0, stream>>>((const float*)regC, (float*)regA);

    // MUSIC spectrum -> specT (zero padded cf lanes first)
    { int n = NB * NAZ * CFP;
      fill_zero_h<<<(n + 255) / 256, 256, 0, stream>>>((_Float16*)regB, n); }
    music_spectrum<<<NB * NF, 128, 0, stream>>>((const float*)regA, svr, svi, (_Float16*)regB);

    // convs2 -> spec2 (B,257,360) f32   (grid y = 6 strips of 64 taps)
    convs2_wmma<<<dim3(17, 6, NB), w32, 0, stream>>>(
        (const _Float16*)regB, wps2, convs2_b, (float*)regC);

    // channel attention
    att_mean<<<(NB * NF + 255) / 256, 256, 0, stream>>>((const float*)regC, yb);
    att_mlp<<<NB, 256, 0, stream>>>(yb, ca_w1, ca_w2, att);

    // final conv + sigmoid -> d_out spectrum, then top-5 soft-argmax -> doa
    convs1_final<<<(NB * NAZ + 255) / 256, 256, 0, stream>>>(
        (const float*)regC, att, convs1_w, convs1_b, outspec);
    top5_doa<<<1, 64, 0, stream>>>(outspec, doa);
}
